// LocalSelfGQA_22694607192163
// MI455X (gfx1250) — matrix operand order repaired and checked by reading, not compiled, not run on hardware
//
#include <hip/hip_runtime.h>
#include <hip/hip_bf16.h>
#include <math.h>
#include <stdint.h>

typedef __bf16 bf16;
typedef __attribute__((ext_vector_type(16))) __bf16 v16bf;
typedef __attribute__((ext_vector_type(8)))  float  v8f;

#define DEV __device__ __forceinline__

static constexpr int B_   = 2;
static constexpr int L_   = 4096;
static constexpr int DIN_ = 1024;
static constexpr int G_   = 8, H_ = 8, DQ_ = 64, DV_ = 64;
static constexpr int S_   = 128;            // 2*L/W query chunks of 32
static constexpr int QDIM = G_ * H_ * DQ_;  // 4096
static constexpr int KVDIM= G_ * (DQ_+DV_); // 1024
static constexpr int FF_  = 1024;
static constexpr int FF2  = 2 * FF_;        // 2048
static constexpr float NEG = -1e30f;

union Frag { v16bf v; uint4 u[2]; };

DEV uint4 ldg16(const bf16* p) { return *(const uint4*)p; }

DEV v8f wmma_bf16(v16bf a, v16bf b, v8f c) {
  // D = A(16x32 bf16) * B(32x16 bf16) + C(16x16 f32), wave32
  return __builtin_amdgcn_wmma_f32_16x16x32_bf16(false, a, false, b, (short)0, c,
                                                 false, false);
}

DEV v8f vzero8() { v8f z = {0.f,0.f,0.f,0.f,0.f,0.f,0.f,0.f}; return z; }

DEV float silu(float x) { return x / (1.0f + __expf(-x)); }

// CDNA5 async global->LDS copy of 16 bytes (tracked by ASYNCcnt).
// lds_byte: LDS byte address (low 32 bits of generic shared pointer).
DEV void async_cp16(unsigned lds_byte, const void* gaddr) {
  asm volatile("global_load_async_to_lds_b128 %0, %1, off"
               :: "v"(lds_byte), "v"(gaddr) : "memory");
}
DEV void wait_async0() { asm volatile("s_wait_asynccnt 0" ::: "memory"); }
DEV void wait_async4() { asm volatile("s_wait_asynccnt 4" ::: "memory"); }

DEV unsigned lds_addr(const void* p) { return (unsigned)(uintptr_t)p; }

// ---------------------------------------------------------------- weight cvt
__global__ __launch_bounds__(256)
void cvt_f32_bf16(const float* __restrict__ src, bf16* __restrict__ dst, size_t n) {
  size_t stride = (size_t)gridDim.x * blockDim.x;
  for (size_t i = (size_t)blockIdx.x * blockDim.x + threadIdx.x; i < n; i += stride)
    dst[i] = (bf16)src[i];
}

// ---------------------------------------------------------------- rmsnorm
DEV float block_sum(float v, float* red) {
  #pragma unroll
  for (int off = 16; off > 0; off >>= 1) v += __shfl_xor(v, off, 32);
  const int lane = threadIdx.x & 31, wave = threadIdx.x >> 5;
  if (lane == 0) red[wave] = v;
  __syncthreads();
  float t = 0.f;
  #pragma unroll
  for (int i = 0; i < 8; ++i) t += red[i];
  __syncthreads();
  return t;
}

__global__ __launch_bounds__(256)
void rmsnorm_f32_to_bf16(const float* __restrict__ x, const float* __restrict__ g,
                         bf16* __restrict__ out, int d) {
  __shared__ float red[8];
  const size_t row = blockIdx.x;
  const float* xr = x + row * (size_t)d;
  float ss = 0.f;
  for (int e = threadIdx.x; e < d; e += 256) { float v = xr[e]; ss += v * v; }
  float tot = block_sum(ss, red);
  float scale = sqrtf((float)d) / fmaxf(sqrtf(tot), 1e-12f);
  for (int e = threadIdx.x; e < d; e += 256)
    out[row * (size_t)d + e] = (bf16)(xr[e] * scale * g[e]);
}

__global__ __launch_bounds__(256)
void rmsnorm_bf16_inplace(bf16* __restrict__ x, const float* __restrict__ g, int d) {
  __shared__ float red[8];
  const size_t row = blockIdx.x;
  bf16* xr = x + row * (size_t)d;
  float ss = 0.f;
  for (int e = threadIdx.x; e < d; e += 256) { float v = (float)xr[e]; ss += v * v; }
  float tot = block_sum(ss, red);
  float scale = sqrtf((float)d) / fmaxf(sqrtf(tot), 1e-12f);
  for (int e = threadIdx.x; e < d; e += 256)
    xr[e] = (bf16)((float)xr[e] * scale * g[e]);
}

// ---------------------------------------------------------------- GEMM
// out[M,N] = act(A[M,K](bf16) * Wt[N,K](bf16)^T + bias[N])
// Workgroup tile 128(M) x 128(N); 8 waves in 4(M) x 2(N); wave tile 32x64.
// Per k-step (32): 128x32 A-panel and 128x32 B-panel double-buffered in LDS
// via global_load_async_to_lds_b128 (ASYNCcnt), overlapped with 8 WMMAs/wave.
template<int ACT, typename OutT>
__global__ __launch_bounds__(256)
void gemm_bf16_wmma(const bf16* __restrict__ A, const bf16* __restrict__ Bt,
                    const float* __restrict__ bias, OutT* __restrict__ out,
                    int M, int N, int K) {
  __shared__ __align__(16) bf16 Asm[2][128 * 32];
  __shared__ __align__(16) bf16 Bsm[2][128 * 32];

  const int tid  = threadIdx.x;
  const int lane = tid & 31;
  const int wave = tid >> 5;
  const int half = lane >> 4;
  const int mrow = lane & 15;
  const int wm   = wave & 3;   // 0..3 -> M sub-tile of 32
  const int wn   = wave >> 2;  // 0..1 -> N sub-tile of 64

  const int mBlock = blockIdx.y * 128;
  const int nBlock = blockIdx.x * 128;

  // staging assignment: 2 x 16B chunks of A-panel + 2 of B-panel per thread
  const int srow  = tid >> 2;          // 0..63
  const int scol8 = (tid & 3) * 8;     // 0,8,16,24

  v8f acc[2][4];
  #pragma unroll
  for (int mt = 0; mt < 2; ++mt)
    #pragma unroll
    for (int nt = 0; nt < 4; ++nt) acc[mt][nt] = vzero8();

  const int nk = K >> 5;

  // prologue: stage k-step 0 into buffer 0
  {
    const int k0 = 0;
    #pragma unroll
    for (int j = 0; j < 2; ++j) {
      const int row = srow + j * 64;
      async_cp16(lds_addr(&Asm[0][row * 32 + scol8]),
                 A  + (size_t)(mBlock + row) * K + k0 + scol8);
      async_cp16(lds_addr(&Bsm[0][row * 32 + scol8]),
                 Bt + (size_t)(nBlock + row) * K + k0 + scol8);
    }
  }

  for (int i = 0; i < nk; ++i) {
    const int db = i & 1;
    if (i + 1 < nk) {
      const int k0 = (i + 1) << 5;
      const int nb = db ^ 1;
      #pragma unroll
      for (int j = 0; j < 2; ++j) {
        const int row = srow + j * 64;
        async_cp16(lds_addr(&Asm[nb][row * 32 + scol8]),
                   A  + (size_t)(mBlock + row) * K + k0 + scol8);
        async_cp16(lds_addr(&Bsm[nb][row * 32 + scol8]),
                   Bt + (size_t)(nBlock + row) * K + k0 + scol8);
      }
      wait_async4();   // the 4 older (stage-i) copies are complete
    } else {
      wait_async0();
    }
    __syncthreads();   // stage-i panels visible to all waves

    const bf16* ap = &Asm[db][0];
    const bf16* bp = &Bsm[db][0];
    Frag fa[2];
    #pragma unroll
    for (int mt = 0; mt < 2; ++mt) {
      const bf16* row = ap + (wm * 32 + mt * 16 + mrow) * 32;
      fa[mt].u[0] = *(const uint4*)(row + half * 8);
      fa[mt].u[1] = *(const uint4*)(row + 16 + half * 8);
    }
    #pragma unroll
    for (int nt = 0; nt < 4; ++nt) {
      Frag fb;
      const bf16* row = bp + (wn * 64 + nt * 16 + mrow) * 32;
      fb.u[0] = *(const uint4*)(row + half * 8);
      fb.u[1] = *(const uint4*)(row + 16 + half * 8);
      #pragma unroll
      for (int mt = 0; mt < 2; ++mt)
        acc[mt][nt] = wmma_bf16(fa[mt].v, fb.v, acc[mt][nt]);
    }
    __syncthreads();   // done reading stage-i panels before they are re-staged
  }

  #pragma unroll
  for (int nt = 0; nt < 4; ++nt) {
    const int col = nBlock + wn * 64 + nt * 16 + mrow;
    const float bv = bias[col];
    #pragma unroll
    for (int mt = 0; mt < 2; ++mt) {
      #pragma unroll
      for (int r = 0; r < 8; ++r) {
        float v = acc[mt][nt][r] + bv;
        if (ACT) v = silu(v);
        out[(size_t)(mBlock + wm * 32 + mt * 16 + half * 8 + r) * N + col] = (OutT)v;
      }
    }
  }
}

// ---------------------------------------------------------------- attention
// One workgroup per (b, g, s-chunk); 8 waves = 8 heads.
// q chunk: 32 rows; k/v window: 64 positions t = s*32 + j - 16 (pad -> masked).
__global__ __launch_bounds__(256)
void local_gqa_attn(const bf16* __restrict__ qb, const bf16* __restrict__ kvb,
                    bf16* __restrict__ ob) {
  __shared__ __align__(16) bf16 Klds[64 * 64];      // [j][d]
  __shared__ __align__(16) bf16 Vtlds[64 * 64];     // [d][j]
  __shared__ __align__(16) bf16 Plds[8 * 32 * 64];  // per wave [m][j]

  const int bid = blockIdx.x;
  const int s   = bid % S_;
  const int g   = (bid / S_) % G_;
  const int b   = bid / (S_ * G_);

  const int tid  = threadIdx.x;
  const int lane = tid & 31;
  const int wave = tid >> 5;          // head
  const int half = lane >> 4;
  const int mrow = lane & 15;

  // ---- stage K (row major) and V (transposed) windows into LDS
  for (int it = 0; it < 2; ++it) {
    const int chunk = tid + it * 256;        // 512 chunks of 8 bf16
    const int j  = chunk >> 3;
    const int d0 = (chunk & 7) * 8;
    const int t  = s * 32 + j - 16;
    const bool ok = (t >= 0) && (t < L_);
    uint4 kv4 = make_uint4(0u, 0u, 0u, 0u);
    if (ok) kv4 = ldg16(kvb + ((size_t)b * L_ + t) * KVDIM + g * 64 + d0);
    *(uint4*)&Klds[j * 64 + d0] = kv4;
    union { uint4 u; bf16 h[8]; } vv;
    vv.u = make_uint4(0u, 0u, 0u, 0u);
    if (ok) vv.u = ldg16(kvb + ((size_t)b * L_ + t) * KVDIM + 512 + g * 64 + d0);
    #pragma unroll
    for (int i = 0; i < 8; ++i) Vtlds[(d0 + i) * 64 + j] = vv.h[i];
  }
  __syncthreads();

  // ---- load Q fragments (A operand), 2 m-tiles x 2 k-steps
  const int qc = (g * H_ + wave) * DQ_;
  Frag aq[2][2];
  #pragma unroll
  for (int mt = 0; mt < 2; ++mt) {
    const bf16* row = qb + ((size_t)b * L_ + s * 32 + mt * 16 + mrow) * QDIM + qc;
    #pragma unroll
    for (int ks = 0; ks < 2; ++ks) {
      aq[mt][ks].u[0] = ldg16(row + ks * 32 + half * 8);
      aq[mt][ks].u[1] = ldg16(row + ks * 32 + 16 + half * 8);
    }
  }

  // ---- S = Q K^T / sqrt(DQ) with edge mask
  v8f sc[2][4];
  #pragma unroll
  for (int mt = 0; mt < 2; ++mt)
    #pragma unroll
    for (int nt = 0; nt < 4; ++nt) sc[mt][nt] = vzero8();

  #pragma unroll
  for (int mt = 0; mt < 2; ++mt)
    #pragma unroll
    for (int nt = 0; nt < 4; ++nt)
      #pragma unroll
      for (int ks = 0; ks < 2; ++ks) {
        Frag bk;
        const int base = (nt * 16 + mrow) * 64 + ks * 32 + half * 8;
        bk.u[0] = *(const uint4*)&Klds[base];
        bk.u[1] = *(const uint4*)&Klds[base + 16];
        sc[mt][nt] = wmma_bf16(aq[mt][ks].v, bk.v, sc[mt][nt]);
      }

  #pragma unroll
  for (int nt = 0; nt < 4; ++nt) {
    const int j = nt * 16 + mrow;
    const int t = s * 32 + j - 16;
    const bool valid = (t >= 0) && (t < L_);
    #pragma unroll
    for (int mt = 0; mt < 2; ++mt)
      #pragma unroll
      for (int r = 0; r < 8; ++r)
        sc[mt][nt][r] = valid ? sc[mt][nt][r] * 0.125f : NEG;
  }

  // ---- softmax along j (16 lanes/half * 4 n-tiles), then P -> LDS (bf16)
  bf16* pbase = &Plds[wave * 32 * 64];
  #pragma unroll
  for (int mt = 0; mt < 2; ++mt)
    #pragma unroll
    for (int r = 0; r < 8; ++r) {
      float mx = NEG;
      #pragma unroll
      for (int nt = 0; nt < 4; ++nt) mx = fmaxf(mx, sc[mt][nt][r]);
      #pragma unroll
      for (int off = 8; off > 0; off >>= 1) mx = fmaxf(mx, __shfl_xor(mx, off, 32));
      float sum = 0.f;
      #pragma unroll
      for (int nt = 0; nt < 4; ++nt) {
        float pe = __expf(sc[mt][nt][r] - mx);
        sc[mt][nt][r] = pe;
        sum += pe;
      }
      #pragma unroll
      for (int off = 8; off > 0; off >>= 1) sum += __shfl_xor(sum, off, 32);
      const float inv = 1.0f / sum;
      #pragma unroll
      for (int nt = 0; nt < 4; ++nt) {
        const float pn = sc[mt][nt][r] * inv;
        pbase[(mt * 16 + half * 8 + r) * 64 + nt * 16 + mrow] = (bf16)pn;
      }
    }
  __syncthreads();

  // ---- O = P V  (A = P from LDS, B = V^T rows from LDS)
  v8f oc[2][4];
  #pragma unroll
  for (int mt = 0; mt < 2; ++mt)
    #pragma unroll
    for (int nt = 0; nt < 4; ++nt) oc[mt][nt] = vzero8();

  #pragma unroll
  for (int mt = 0; mt < 2; ++mt)
    #pragma unroll
    for (int ks = 0; ks < 2; ++ks) {
      Frag ap;
      const bf16* prow = pbase + (mt * 16 + mrow) * 64;
      ap.u[0] = *(const uint4*)(prow + ks * 32 + half * 8);
      ap.u[1] = *(const uint4*)(prow + ks * 32 + 16 + half * 8);
      #pragma unroll
      for (int nt = 0; nt < 4; ++nt) {
        Frag bv;
        const int base = (nt * 16 + mrow) * 64 + ks * 32 + half * 8;
        bv.u[0] = *(const uint4*)&Vtlds[base];
        bv.u[1] = *(const uint4*)&Vtlds[base + 16];
        oc[mt][nt] = wmma_bf16(ap.v, bv.v, oc[mt][nt]);
      }
    }

  #pragma unroll
  for (int mt = 0; mt < 2; ++mt)
    #pragma unroll
    for (int nt = 0; nt < 4; ++nt) {
      const int col = qc + nt * 16 + mrow;
      #pragma unroll
      for (int r = 0; r < 8; ++r) {
        const int t = s * 32 + mt * 16 + half * 8 + r;
        ob[((size_t)b * L_ + t) * QDIM + col] = (bf16)oc[mt][nt][r];
      }
    }
}

// ---------------------------------------------------------------- launch
extern "C" void kernel_launch(void* const* d_in, const int* in_sizes, int n_in,
                              void* d_out, int out_size, void* d_ws, size_t ws_size,
                              hipStream_t stream) {
  const float* x      = (const float*)d_in[0];
  const float* ngamma = (const float*)d_in[1];
  const float* Wq_w   = (const float*)d_in[2];
  const float* Wq_b   = (const float*)d_in[3];
  const float* Wkv_w  = (const float*)d_in[4];
  const float* Wkv_b  = (const float*)d_in[5];
  const float* g1     = (const float*)d_in[6];
  const float* W1_w   = (const float*)d_in[7];
  const float* W1_b   = (const float*)d_in[8];
  const float* g2     = (const float*)d_in[9];
  const float* W2_w   = (const float*)d_in[10];
  const float* W2_b   = (const float*)d_in[11];
  float* out = (float*)d_out;

  const int M = B_ * L_;  // 8192 rows everywhere

  char* p = (char*)d_ws;
  bf16* Wq_bf  = (bf16*)p; p += (size_t)QDIM  * DIN_ * 2;
  bf16* Wkv_bf = (bf16*)p; p += (size_t)KVDIM * DIN_ * 2;
  bf16* W1_bf  = (bf16*)p; p += (size_t)FF2   * QDIM * 2;
  bf16* W2_bf  = (bf16*)p; p += (size_t)FF_   * FF2  * 2;
  bf16* xn     = (bf16*)p; p += (size_t)M * DIN_  * 2;
  bf16* qbuf   = (bf16*)p; p += (size_t)M * QDIM  * 2;   // q, then o (aliased)
  bf16* kvbuf  = (bf16*)p; p += (size_t)M * KVDIM * 2;
  bf16* z1     = (bf16*)p; p += (size_t)M * FF2   * 2;

  // weights -> bf16
  cvt_f32_bf16<<<2048, 256, 0, stream>>>(Wq_w,  Wq_bf,  (size_t)QDIM  * DIN_);
  cvt_f32_bf16<<<2048, 256, 0, stream>>>(Wkv_w, Wkv_bf, (size_t)KVDIM * DIN_);
  cvt_f32_bf16<<<2048, 256, 0, stream>>>(W1_w,  W1_bf,  (size_t)FF2   * QDIM);
  cvt_f32_bf16<<<2048, 256, 0, stream>>>(W2_w,  W2_bf,  (size_t)FF_   * FF2);

  // xn = rmsnorm(x) (bf16)
  rmsnorm_f32_to_bf16<<<M, 256, 0, stream>>>(x, ngamma, xn, DIN_);

  // q / kv projections
  gemm_bf16_wmma<0, bf16><<<dim3(QDIM / 128, M / 128), 256, 0, stream>>>(
      xn, Wq_bf, Wq_b, qbuf, M, QDIM, DIN_);
  gemm_bf16_wmma<0, bf16><<<dim3(KVDIM / 128, M / 128), 256, 0, stream>>>(
      xn, Wkv_bf, Wkv_b, kvbuf, M, KVDIM, DIN_);

  // windowed GQA attention (o overwrites q in place, chunk-locally)
  local_gqa_attn<<<B_ * G_ * S_, 256, 0, stream>>>(qbuf, kvbuf, qbuf);

  // z = silu(rmsnorm(o) W1^T + b1)
  rmsnorm_bf16_inplace<<<M, 256, 0, stream>>>(qbuf, g1, QDIM);
  gemm_bf16_wmma<1, bf16><<<dim3(FF2 / 128, M / 128), 256, 0, stream>>>(
      qbuf, W1_bf, W1_b, z1, M, FF2, QDIM);

  // out = silu(rmsnorm(z) W2^T + b2) (f32)
  rmsnorm_bf16_inplace<<<M, 256, 0, stream>>>(z1, g2, FF2);
  gemm_bf16_wmma<1, float><<<dim3(FF_ / 128, M / 128), 256, 0, stream>>>(
      z1, W2_bf, W2_b, out, M, FF_, FF2);
}
